// CrossAttention_17145509446496
// MI455X (gfx1250) — matrix operand order repaired and checked by reading, not compiled, not run on hardware
//
#include <hip/hip_runtime.h>

// ---------------------------------------------------------------------------
// Fused cross-attention for MI455X (gfx1250, wave32, WMMA).
//   B=2, N=1024, C=64, H=8, d=8, K=LF=256, F=2*LF=512
// nb = neighbors @ W_two^T  -> bf16 WMMA (dominant 34 GFLOP, HBM-bound)
// k,v never materialized: folded into per-point qw vectors / Wv epilogue.
// ---------------------------------------------------------------------------

typedef __attribute__((ext_vector_type(16))) __bf16 v16bf;
typedef __attribute__((ext_vector_type(8)))  float  v8f;
typedef __attribute__((ext_vector_type(4)))  float  v4f;

#define NPTS      1024
#define NB_STRIDE 260     // 256 + 4 pad: kills LDS bank conflicts on WMMA store

// ---- prep 1: W_two (256x512 f32, row-major) -> bf16 same layout -----------
__global__ void prep_wtwo_kernel(const float* __restrict__ W_two,
                                 __bf16* __restrict__ dst) {
    int i = blockIdx.x * blockDim.x + threadIdx.x;   // 131072 elems exactly
    dst[i] = (__bf16)W_two[i];
}

// ---- prep 2: qw[b,n,h,c] = (1/sqrt(d)) * sum_i (pcd[b,n,:]@Wq^T)[h*8+i] * Wk[h*8+i,c]
__global__ void prep_qw_kernel(const float* __restrict__ pcd,
                               const float* __restrict__ Wq,
                               const float* __restrict__ Wk,
                               float* __restrict__ qw) {
    __shared__ float qsh[64];
    const int idx = blockIdx.x;          // b*N + n  (2048 blocks of 64 threads)
    const int t   = threadIdx.x;         // 0..63
    const float* p = pcd + (size_t)idx * 64;
    float acc = 0.f;
    #pragma unroll 8
    for (int c = 0; c < 64; ++c) acc = __builtin_fmaf(p[c], Wq[t * 64 + c], acc);
    qsh[t] = acc;                        // q[o] for o = t
    __syncthreads();
    const float scale = 0.35355339059327373f;  // 1/sqrt(8)
    #pragma unroll
    for (int h = 0; h < 8; ++h) {
        float s = 0.f;
        #pragma unroll
        for (int i = 0; i < 8; ++i)
            s = __builtin_fmaf(qsh[h * 8 + i], Wk[(h * 8 + i) * 64 + t], s);
        qw[((size_t)idx * 8 + h) * 64 + t] = s * scale;
    }
}

// ---- main fused kernel: one workgroup (8 waves) per point (b,n) -----------
__launch_bounds__(256)
__global__ void fused_attn_kernel(const float*  __restrict__ neighbors, // (2,64,1024,512)
                                  const float*  __restrict__ b_two,     // (256,)
                                  const float*  __restrict__ Wv,        // (64,64)
                                  const __bf16* __restrict__ wtwo_bf,   // (256,512) bf16
                                  const float*  __restrict__ qw,        // (B*N,8,64)
                                  float*        __restrict__ out) {     // (2,64,1024)
    __shared__ float nb[64 * NB_STRIDE];   // nb[c][kk] f32   (66.5 KB)
    __shared__ float sc[8 * 256];          // scores -> attn  (8 KB)
    __shared__ float qwl[8 * 64];          // per-point qw    (2 KB)
    __shared__ float avl[8 * 64];          // attn @ nb^T     (2 KB)

    const int idx  = blockIdx.x;           // b*N + n
    const int b    = idx >> 10;
    const int n    = idx & (NPTS - 1);
    const int tid  = threadIdx.x;
    const int wave = tid >> 5;
    const int lane = tid & 31;
    const int lrow = lane & 15;
    const int hi   = lane >> 4;            // lane half selector

    // ---------------- Phase 1: nb = A(64x512) x W_two^T(512x256), bf16 WMMA
    // wave -> c-tile ct = wave>>1 (rows ct*16..+15), kk-half kh = wave&1
    const int ct = wave >> 1;
    const int kh = wave & 1;
    const int c  = ct * 16 + lrow;
    const float* arow = neighbors + ((size_t)(b * 64 + c) * NPTS + n) * 512;

    v8f acc[8] = {};                       // 8 kk-tiles of 16x16 f32
    for (int f0 = 0; f0 < 512; f0 += 32) {
        // A fragment 16x32 bf16: lanes<16 hold K f0+[0..15]; lanes>=16 +16
        const float* p = arow + f0 + hi * 16;
        v4f a0 = *(const v4f*)(p);
        v4f a1 = *(const v4f*)(p + 4);
        v4f a2 = *(const v4f*)(p + 8);
        v4f a3 = *(const v4f*)(p + 12);
        v16bf afrag;
        #pragma unroll
        for (int e = 0; e < 4; ++e) {
            afrag[e]      = (__bf16)a0[e];
            afrag[e + 4]  = (__bf16)a1[e];
            afrag[e + 8]  = (__bf16)a2[e];
            afrag[e + 12] = (__bf16)a3[e];
        }
        #pragma unroll
        for (int t = 0; t < 8; ++t) {
            // B fragment 32x16: lane col = kk, lanes<16 K f0+[0..15], lanes>=16 +16
            const int col = (kh * 8 + t) * 16 + lrow;
            const __bf16* bp = wtwo_bf + (size_t)col * 512 + f0 + hi * 16;
            v16bf bfrag = *(const v16bf*)bp;   // 32B, split into 2x b128
            acc[t] = __builtin_amdgcn_wmma_f32_16x16x32_bf16(
                false, afrag, false, bfrag, (short)0, acc[t], false, false);
        }
    }
    // store D tiles to LDS (+ bias): VGPR r -> row ct*16 + r + hi*8, col = lane%16
    #pragma unroll
    for (int t = 0; t < 8; ++t) {
        const int col  = (kh * 8 + t) * 16 + lrow;
        const float bias = b_two[col];
        const int rb   = ct * 16 + hi * 8;
        #pragma unroll
        for (int r = 0; r < 8; ++r)
            nb[(rb + r) * NB_STRIDE + col] = acc[t][r] + bias;
    }
    // stage qw for this point (512 f32)
    qwl[tid]       = qw[(size_t)idx * 512 + tid];
    qwl[tid + 256] = qw[(size_t)idx * 512 + tid + 256];
    __syncthreads();

    // ---------------- Phase 2a: scores[h][kk] = sum_c qw[h][c] * nb[c][kk]
    {
        float s[8] = {0.f, 0.f, 0.f, 0.f, 0.f, 0.f, 0.f, 0.f};
        for (int cc = 0; cc < 64; ++cc) {
            const float nv = nb[cc * NB_STRIDE + tid];   // conflict-free
            #pragma unroll
            for (int h = 0; h < 8; ++h)                  // qwl uniform -> broadcast
                s[h] = __builtin_fmaf(qwl[h * 64 + cc], nv, s[h]);
        }
        #pragma unroll
        for (int h = 0; h < 8; ++h) sc[h * 256 + tid] = s[h];
    }
    __syncthreads();

    // ---------------- Phase 2b: softmax over kk, wave `wave` owns head h=wave
    {
        const int h = wave;
        float v[8];
        float m = -3.4e38f;
        #pragma unroll
        for (int j = 0; j < 8; ++j) {
            v[j] = sc[h * 256 + j * 32 + lane];
            m = fmaxf(m, v[j]);
        }
        #pragma unroll
        for (int off = 16; off > 0; off >>= 1) m = fmaxf(m, __shfl_xor(m, off, 32));
        float sum = 0.f;
        #pragma unroll
        for (int j = 0; j < 8; ++j) { v[j] = __expf(v[j] - m); sum += v[j]; }
        #pragma unroll
        for (int off = 16; off > 0; off >>= 1) sum += __shfl_xor(sum, off, 32);
        const float inv = __frcp_rn(sum);
        #pragma unroll
        for (int j = 0; j < 8; ++j) sc[h * 256 + j * 32 + lane] = v[j] * inv;
    }
    __syncthreads();

    // ---------------- Phase 2c: av[h][c] = sum_kk attn[h][kk] * nb[c][kk]
    {
        const int h  = wave;
        const int c1 = lane;               // c1 and c1+32 -> only 2-way LDS conflict
        float a0 = 0.f, a1 = 0.f;
        for (int kk = 0; kk < 256; ++kk) {
            const float at = sc[h * 256 + kk];          // wave-uniform broadcast
            a0 = __builtin_fmaf(at, nb[c1 * NB_STRIDE + kk], a0);
            a1 = __builtin_fmaf(at, nb[(c1 + 32) * NB_STRIDE + kk], a1);
        }
        avl[h * 64 + c1]      = a0;
        avl[h * 64 + c1 + 32] = a1;
    }
    __syncthreads();

    // ---------------- Phase 2d: out[b, o, n] = sum_c Wv[o][c] * av[o/8][c]
    if (tid < 64) {
        const int o = tid, h = o >> 3;
        float x = 0.f;
        #pragma unroll 8
        for (int cc = 0; cc < 64; ++cc)
            x = __builtin_fmaf(Wv[o * 64 + cc], avl[h * 64 + cc], x);
        out[((size_t)b * 64 + o) * NPTS + n] = x;
    }
}

// ---------------------------------------------------------------------------
extern "C" void kernel_launch(void* const* d_in, const int* in_sizes, int n_in,
                              void* d_out, int out_size, void* d_ws, size_t ws_size,
                              hipStream_t stream) {
    const float* pcd       = (const float*)d_in[0];  // (2,1024,64)
    const float* neighbors = (const float*)d_in[1];  // (2,64,1024,512)
    const float* W_two     = (const float*)d_in[2];  // (256,512)
    const float* b_two     = (const float*)d_in[3];  // (256,)
    const float* Wq        = (const float*)d_in[4];  // (64,64)
    const float* Wk        = (const float*)d_in[5];  // (64,64)
    const float* Wv        = (const float*)d_in[6];  // (64,64)
    float* out = (float*)d_out;                      // (2,64,1024)

    // workspace: [0,256KB) bf16 W_two ; [256KB, 256KB+4MB) qw f32
    __bf16* wtwo_bf = (__bf16*)d_ws;
    float*  qw      = (float*)((char*)d_ws + (size_t)256 * 512 * 2);

    prep_wtwo_kernel<<<512, 256, 0, stream>>>(W_two, wtwo_bf);
    prep_qw_kernel<<<2048, 64, 0, stream>>>(pcd, Wq, Wk, qw);
    fused_attn_kernel<<<2048, 256, 0, stream>>>(neighbors, b_two, Wv, wtwo_bf, qw, out);
}
